// HyperedgeGenerator_17549236371597
// MI455X (gfx1250) — matrix operand order repaired and checked by reading, not compiled, not run on hardware
//
#include <hip/hip_runtime.h>
#include <hip/hip_bf16.h>

typedef __attribute__((ext_vector_type(16))) _Float16 v16h;
typedef __attribute__((ext_vector_type(8)))  _Float16 v8h;
typedef __attribute__((ext_vector_type(8)))  float    v8f;
typedef __attribute__((ext_vector_type(4)))  float    v4f;

#define N_NODES 8192
#define HIDDEN  256
#define TOPK    16
#define THRESH  0.5f

// sim kernel tiling
#define MTILE   32      // rows per block
#define NTILE   128     // columns per sweep step (8 waves x 16)

#define WMMA_F32_F16(a, b, c) \
  __builtin_amdgcn_wmma_f32_16x16x32_f16(false, (a), false, (b), (short)0, (c), false, false)

// ---------------- fragment loaders (CDNA5 16-bit WMMA layouts) ----------------
// A 16x32 f16: lanes 0-15 row=lane, halves 0..7 = K k..k+7, halves 8..15 = K k+16..k+23;
// lanes 16-31 same row, K offset +8.
__device__ __forceinline__ v16h load_a_frag_f32(const float* __restrict__ x, int ld,
                                                int row, int k, int lane) {
  const int off = (lane & 16) ? 8 : 0;
  const float* p0 = x + (size_t)row * ld + (k + off);
  v4f q0 = *(const v4f*)(p0);
  v4f q1 = *(const v4f*)(p0 + 4);
  v4f q2 = *(const v4f*)(p0 + 16);
  v4f q3 = *(const v4f*)(p0 + 20);
  v16h a;
#pragma unroll
  for (int i = 0; i < 4; ++i) {
    a[i]      = (_Float16)q0[i];
    a[4 + i]  = (_Float16)q1[i];
    a[8 + i]  = (_Float16)q2[i];
    a[12 + i] = (_Float16)q3[i];
  }
  return a;
}

// A fragment from LDS-resident row-major f16 tile (ld halves per row)
__device__ __forceinline__ v16h load_a_frag_lds(const _Float16* __restrict__ base, int ld,
                                                int row, int k, int lane) {
  const int off = (lane & 16) ? 8 : 0;
  const _Float16* p = base + row * ld + (k + off);
  v8h lo = *(const v8h*)(p);        // ds_read_b128
  v8h hi = *(const v8h*)(p + 16);   // ds_read_b128
  v16h a;
#pragma unroll
  for (int i = 0; i < 8; ++i) { a[i] = lo[i]; a[8 + i] = hi[i]; }
  return a;
}

// B 32x16 f16: lanes 0-15 col=lane hold K k..k+7, k+16..k+23; lanes 16-31 hold K k+8..k+15, k+24..k+31.
// wt is (N x K) row-major, so a lane reads two runs of 8 contiguous halves (k+koff and k+koff+16).
__device__ __forceinline__ v16h load_b_frag_f16(const _Float16* __restrict__ wt, int ld,
                                                int col, int k, int lane) {
  const int koff = (lane & 16) ? 8 : 0;
  const _Float16* p = wt + (size_t)col * ld + (k + koff);
  v8h lo = *(const v8h*)(p);
  v8h hi = *(const v8h*)(p + 16);
  v16h b;
#pragma unroll
  for (int i = 0; i < 8; ++i) { b[i] = lo[i]; b[8 + i] = hi[i]; }
  return b;
}

// Register-resident sorted (descending) top-16 insert; fully unrolled -> no scratch.
__device__ __forceinline__ void topk_insert(float (&v)[TOPK], int (&id)[TOPK],
                                            float c, int ci) {
  if (c <= v[TOPK - 1]) return;
  float cv = c; int cidx = ci;
#pragma unroll
  for (int j = 0; j < TOPK; ++j) {
    if (cv > v[j]) {
      float tv = v[j]; int ti = id[j];
      v[j] = cv; id[j] = cidx;
      cv = tv; cidx = ti;
    }
  }
}

// ---------------- W convert + transpose: (K x 256) f32 -> (256 x K) f16 ----------------
__global__ __launch_bounds__(256) void wcvt_kernel(const float* __restrict__ w,
                                                   _Float16* __restrict__ wt, int K) {
  int idx = blockIdx.x * blockDim.x + threadIdx.x;
  if (idx < K * HIDDEN) {
    int k = idx >> 8;        // HIDDEN == 256
    int n = idx & 255;
    wt[(size_t)n * K + k] = (_Float16)w[idx];
  }
}

// ---------------- zero-fill H ----------------
__global__ __launch_bounds__(256) void zero_kernel(v4f* __restrict__ out, long n4) {
  long i = (long)blockIdx.x * blockDim.x + threadIdx.x;
  if (i < n4) { v4f z = {0.f, 0.f, 0.f, 0.f}; out[i] = z; }
}

// ---------------- fused 3-modality encoder: fused = mean(relu(x W + b)), hn = fused/(|fused|+eps) ----------------
__global__ __launch_bounds__(256) void encode_kernel(
    const float* __restrict__ x0, const _Float16* __restrict__ wt0, const float* __restrict__ bb0, int k0,
    const float* __restrict__ x1, const _Float16* __restrict__ wt1, const float* __restrict__ bb1, int k1,
    const float* __restrict__ x2, const _Float16* __restrict__ wt2, const float* __restrict__ bb2, int k2,
    _Float16* __restrict__ hn) {
  __shared__ float fused_s[16][HIDDEN + 4];
  __shared__ float partial_s[16][16];

  const int tid = threadIdx.x;
  const int lane = tid & 31;
  const int wave = tid >> 5;
  const int rowBase = blockIdx.x * 16;
  const int colSub = lane & 15;
  const int aRow = rowBase + colSub;

  const float*    xs[3]  = {x0, x1, x2};
  const _Float16* wts[3] = {wt0, wt1, wt2};
  const float*    bs[3]  = {bb0, bb1, bb2};
  const int       ks[3]  = {k0, k1, k2};

  const int col0 = (wave * 2) * 16 + colSub;
  const int col1 = col0 + 16;

  v8f acc0 = {}; v8f acc1 = {};

#pragma unroll
  for (int m = 0; m < 3; ++m) {
    const float* x = xs[m];
    const _Float16* wt = wts[m];
    const int K = ks[m];
    v8f c0 = {}; v8f c1 = {};
    for (int k = 0; k < K; k += 32) {
      v16h a   = load_a_frag_f32(x, K, aRow, k, lane);
      v16h fb0 = load_b_frag_f16(wt, K, col0, k, lane);
      v16h fb1 = load_b_frag_f16(wt, K, col1, k, lane);
      c0 = WMMA_F32_F16(a, fb0, c0);
      c1 = WMMA_F32_F16(a, fb1, c1);
    }
    const float bias0 = bs[m][col0];
    const float bias1 = bs[m][col1];
#pragma unroll
    for (int j = 0; j < 8; ++j) {
      float t0 = c0[j] + bias0; acc0[j] += (t0 > 0.f) ? t0 : 0.f;
      float t1 = c1[j] + bias1; acc1[j] += (t1 > 0.f) ? t1 : 0.f;
    }
  }

  // C layout: VGPR j -> row j (+8 for lanes 16-31), col = lane&15 within tile.
  const int rofs = (lane & 16) ? 8 : 0;
#pragma unroll
  for (int j = 0; j < 8; ++j) {
    fused_s[j + rofs][col0] = acc0[j] * (1.f / 3.f);
    fused_s[j + rofs][col1] = acc1[j] * (1.f / 3.f);
  }
  __syncthreads();

  const int r  = tid >> 4;   // 0..15
  const int cs = tid & 15;
  float ss = 0.f;
#pragma unroll
  for (int i = 0; i < HIDDEN / 16; ++i) {
    float v = fused_s[r][cs + i * 16];
    ss += v * v;
  }
  partial_s[r][cs] = ss;
  __syncthreads();

  float sum = 0.f;
#pragma unroll
  for (int i = 0; i < 16; ++i) sum += partial_s[r][i];
  const float inv = 1.0f / (sqrtf(sum) + 1e-8f);
#pragma unroll
  for (int i = 0; i < HIDDEN / 16; ++i) {
    int c = cs + i * 16;
    hn[(size_t)(rowBase + r) * HIDDEN + c] = (_Float16)(fused_s[r][c] * inv);
  }
}

// ---------------- sim = hn hn^T swept in tiles, fused top-16 + threshold + scatter ----------------
// Block: 32 rows (2 WMMA row tiles, staged in LDS), sweep N columns in 128-wide steps.
// Each of 8 waves owns one 16-col tile and both row tiles -> each B fragment feeds 2 WMMAs.
__global__ __launch_bounds__(256) void sim_topk_kernel(const _Float16* __restrict__ hn,
                                                       float* __restrict__ out) {
  __shared__ _Float16 a_s[MTILE * HIDDEN];        // 16 KB: block's 32 hn rows
  __shared__ float    sim_s[MTILE][NTILE + 4];    // sim tile; reused as cand values
  __shared__ int      cand_i[MTILE][NTILE];

  const int tid = threadIdx.x;
  const int lane = tid & 31;
  const int wave = tid >> 5;
  const int rowBase = blockIdx.x * MTILE;
  const int colSub = lane & 15;

  // ---- stage A rows into LDS (row-major f16 [32][256]) ----
  {
    const v4f* src = (const v4f*)(hn + (size_t)rowBase * HIDDEN); // 16B granules
    v4f* dst = (v4f*)a_s;
    const int n16 = MTILE * HIDDEN / 8;   // # of 16-byte granules = 1024
#pragma unroll
    for (int i = tid; i < n16; i += 256) dst[i] = src[i];
  }
  __syncthreads();

  const int r  = tid >> 3;   // scan row 0..31
  const int cs = tid & 7;    // column residue class mod 8
  const int grow = rowBase + r;

  float bv[TOPK]; int bi[TOPK];
#pragma unroll
  for (int j = 0; j < TOPK; ++j) { bv[j] = -2.0f; bi[j] = 0; }

  const int aRow0 = colSub;        // row tile 0 lane row (LDS-local)
  const int aRow1 = 16 + colSub;   // row tile 1 lane row

  const int ntiles = N_NODES / NTILE;
  for (int tile = 0; tile < ntiles; ++tile) {
    const int colBase = tile * NTILE;
    const int lc = wave * 16 + colSub;     // local column within tile
    const int col = colBase + lc;          // global column

    if (tile + 1 < ntiles)
      __builtin_prefetch(hn + (size_t)(col + NTILE) * HIDDEN, 0, 1);

    v8f c0 = {}, c1 = {};
#pragma unroll
    for (int k = 0; k < HIDDEN; k += 32) {
      v16h fb = load_b_frag_f16(hn, HIDDEN, col, k, lane);
      v16h a0 = load_a_frag_lds(a_s, HIDDEN, aRow0, k, lane);
      v16h a1 = load_a_frag_lds(a_s, HIDDEN, aRow1, k, lane);
      c0 = WMMA_F32_F16(a0, fb, c0);
      c1 = WMMA_F32_F16(a1, fb, c1);
    }

    __syncthreads();   // previous scan done before overwriting sim_s
    const int rofs = (lane & 16) ? 8 : 0;
#pragma unroll
    for (int j = 0; j < 8; ++j) {
      sim_s[j + rofs][lc]      = c0[j];
      sim_s[16 + j + rofs][lc] = c1[j];
    }
    __syncthreads();

    // each thread scans its residue class (disjoint columns across a row's 8 threads)
#pragma unroll
    for (int i = 0; i < NTILE / 8; ++i) {
      int c = cs + i * 8;
      int gcol = colBase + c;
      float v = sim_s[r][c];
      if (gcol == grow) v = -1.0f;   // exclude self, matching reference
      topk_insert(bv, bi, v, gcol);
    }
  }

  // ---- merge 8 per-thread top-16 lists per row ----
  __syncthreads();
#pragma unroll
  for (int j = 0; j < TOPK; ++j) {
    sim_s[r][cs * TOPK + j]  = bv[j];
    cand_i[r][cs * TOPK + j] = bi[j];
  }
  __syncthreads();

  if (tid < MTILE) {
    const int row = tid;
    float tv[TOPK]; int ti[TOPK];
#pragma unroll
    for (int j = 0; j < TOPK; ++j) { tv[j] = -2.0f; ti[j] = 0; }
    for (int j = 0; j < 8 * TOPK; ++j)
      topk_insert(tv, ti, sim_s[row][j], cand_i[row][j]);

    float s = 0.f;
#pragma unroll
    for (int j = 0; j < TOPK; ++j) s += tv[j];
    s *= (1.0f / TOPK);

    if (s > THRESH) {
      float* orow = out + (size_t)(rowBase + row) * N_NODES;
#pragma unroll
      for (int j = 0; j < TOPK; ++j) orow[ti[j]] = tv[j];
    }
  }
}

// ---------------- launch ----------------
extern "C" void kernel_launch(void* const* d_in, const int* in_sizes, int n_in,
                              void* d_out, int out_size, void* d_ws, size_t ws_size,
                              hipStream_t stream) {
  const float* x_img = (const float*)d_in[0];
  const float* W_img = (const float*)d_in[1];
  const float* b_img = (const float*)d_in[2];
  const float* x_txt = (const float*)d_in[3];
  const float* W_txt = (const float*)d_in[4];
  const float* b_txt = (const float*)d_in[5];
  const float* x_aud = (const float*)d_in[6];
  const float* W_aud = (const float*)d_in[7];
  const float* b_aud = (const float*)d_in[8];
  float* out = (float*)d_out;

  char* ws = (char*)d_ws;
  _Float16* wt_img = (_Float16*)ws;  ws += (size_t)HIDDEN * 1024 * 2;
  _Float16* wt_txt = (_Float16*)ws;  ws += (size_t)HIDDEN * 768 * 2;
  _Float16* wt_aud = (_Float16*)ws;  ws += (size_t)HIDDEN * 512 * 2;
  _Float16* hn     = (_Float16*)ws;  // N_NODES * HIDDEN f16

  wcvt_kernel<<<(1024 * HIDDEN + 255) / 256, 256, 0, stream>>>(W_img, wt_img, 1024);
  wcvt_kernel<<<(768  * HIDDEN + 255) / 256, 256, 0, stream>>>(W_txt, wt_txt, 768);
  wcvt_kernel<<<(512  * HIDDEN + 255) / 256, 256, 0, stream>>>(W_aud, wt_aud, 512);

  long n4 = (long)N_NODES * N_NODES / 4;
  zero_kernel<<<(int)((n4 + 255) / 256), 256, 0, stream>>>((v4f*)out, n4);

  encode_kernel<<<N_NODES / 16, 256, 0, stream>>>(
      x_img, wt_img, b_img, 1024,
      x_txt, wt_txt, b_txt, 768,
      x_aud, wt_aud, b_aud, 512, hn);

  sim_topk_kernel<<<N_NODES / MTILE, 256, 0, stream>>>(hn, out);
}
